// CausalSelfAttention_61735859913202
// MI455X (gfx1250) — compile-verified, matrix operand order checked by reading, not run on hardware
//
#include <hip/hip_runtime.h>

// Problem dims
#define B_ 4
#define T_ 2048
#define C_ 1024
#define H_ 16
#define D_ 64

typedef _Float16 h8   __attribute__((ext_vector_type(8)));
typedef _Float16 v16h __attribute__((ext_vector_type(16)));
typedef float    v8f  __attribute__((ext_vector_type(8)));
typedef int      i4g  __attribute__((vector_size(16)));   // matches builtin param type

#define __AS1 __attribute__((address_space(1)))
#define __AS3 __attribute__((address_space(3)))

static __device__ __forceinline__ v16h comb(h8 lo, h8 hi) {
  return __builtin_shufflevector(lo, hi, 0,1,2,3,4,5,6,7,8,9,10,11,12,13,14,15);
}
static __device__ __forceinline__ h8 ldg8(const _Float16* p) {
  return *(const h8*)p;   // 16B aligned -> global_load_b128 / ds_load_b128
}
static __device__ __forceinline__ v8f wmma_f16(v16h a, v16h b, v8f c) {
  // D = A(16x32 f16) * B(32x16 f16) + C(16x16 f32)
  return __builtin_amdgcn_wmma_f32_16x16x32_f16(false, a, false, b, (short)0, c, false, false);
}

// 16B global -> LDS copy: async path when available, else direct store.
static __device__ __forceinline__ void cp16(_Float16* ldst, const _Float16* gsrc) {
#if __has_builtin(__builtin_amdgcn_global_load_async_to_lds_b128)
  __builtin_amdgcn_global_load_async_to_lds_b128(
      (__AS1 i4g*)(gsrc), (__AS3 i4g*)(ldst), 0, 0);
#else
  *(h8*)ldst = *(const h8*)gsrc;
#endif
}

// ---------------- fp32 -> fp16 conversion ----------------
__global__ __launch_bounds__(256) void k_cvt(const float* __restrict__ in,
                                             _Float16* __restrict__ out, int n) {
  int i = blockIdx.x * 256 + threadIdx.x;
  if (i < n) out[i] = (_Float16)in[i];
}

// ---------------- shared GEMM core -----------------------
// Block tile 64(M) x 128(N), K-step 32, double-buffered LDS staging.
// 8 waves: wm = wave&3 (M strips of 16), wn = wave>>2 (N strips of 64).
#define AROWS 64
#define BROWS 128
#define KSTEP 32
#define LPAD  8
#define LSTR  (KSTEP + LPAD)                 // 40 halves per LDS row
#define SA_HALVES (2 * AROWS * LSTR)         //  5120 halves
#define SB_HALVES (2 * BROWS * LSTR)         // 10240 halves

static __device__ __forceinline__ void gemm_issue(
    _Float16* sA, _Float16* sB,
    const _Float16* __restrict__ Ag, const _Float16* __restrict__ Bg,
    int buf, int kk, int tid) {
  {
    const int r = tid >> 2, c = (tid & 3) * 8;               // 64 rows x 4 chunks
    cp16(sA + ((buf * AROWS + r) * LSTR + c), Ag + (long)r * C_ + kk + c);
  }
#pragma unroll
  for (int u = 0; u < 2; ++u) {                              // 128 rows x 4 chunks
    const int t = tid + u * 256;
    const int r = t >> 2, c = (t & 3) * 8;
    cp16(sB + ((buf * BROWS + r) * LSTR + c), Bg + (long)r * C_ + kk + c);
  }
}

static __device__ __forceinline__ void gemm_compute(
    const _Float16* sA, const _Float16* sB, int buf, int tid, v8f acc[4]) {
  const int lane = tid & 31, wid = tid >> 5;
  const int wm = wid & 3, wn = wid >> 2;
  const int n16 = lane & 15, grp = lane >> 4;
  const _Float16* ar = sA + ((buf * AROWS + wm * 16 + n16) * LSTR);
  const _Float16* br = sB + ((buf * BROWS + wn * 64 + n16) * LSTR);
  v16h a  = comb(ldg8(ar + grp * 8),              ldg8(ar + 16 + grp * 8));
  v16h b0 = comb(ldg8(br + grp * 8),              ldg8(br + 16 + grp * 8));
  v16h b1 = comb(ldg8(br + 16 * LSTR + grp * 8),  ldg8(br + 16 * LSTR + 16 + grp * 8));
  v16h b2 = comb(ldg8(br + 32 * LSTR + grp * 8),  ldg8(br + 32 * LSTR + 16 + grp * 8));
  v16h b3 = comb(ldg8(br + 48 * LSTR + grp * 8),  ldg8(br + 48 * LSTR + 16 + grp * 8));
  // Pin: every WMMA depends (through the asm) on ALL fragment loads, forcing
  // one batched ds clause + single s_wait_dscnt instead of per-WMMA waits.
  asm volatile("" : "+v"(a), "+v"(b0), "+v"(b1), "+v"(b2), "+v"(b3));
  acc[0] = wmma_f16(a, b0, acc[0]);
  acc[1] = wmma_f16(a, b1, acc[1]);
  acc[2] = wmma_f16(a, b2, acc[2]);
  acc[3] = wmma_f16(a, b3, acc[3]);
}

static __device__ __forceinline__ void gemm_loop(
    _Float16* sA, _Float16* sB,
    const _Float16* __restrict__ Ag, const _Float16* __restrict__ Bg,
    int tid, v8f acc[4]) {
  gemm_issue(sA, sB, Ag, Bg, 0, 0, tid);
  for (int kt = 0; kt < (C_ / KSTEP) - 1; ++kt) {
    gemm_issue(sA, sB, Ag, Bg, (kt + 1) & 1, (kt + 1) * KSTEP, tid);
    asm volatile("s_wait_asynccnt 0x3" ::: "memory");  // drain older 3-instr batch
    __syncthreads();
    gemm_compute(sA, sB, kt & 1, tid, acc);
    __syncthreads();                                   // protect buffer reuse
  }
  asm volatile("s_wait_asynccnt 0x0" ::: "memory");
  __syncthreads();
  gemm_compute(sA, sB, ((C_ / KSTEP) - 1) & 1, tid, acc);
}

// ---------------- fused QKV GEMM -------------------------
// grid (M/64, N/128, 3); block 256
__global__ __launch_bounds__(256) void k_gemm_qkv(
    const _Float16* __restrict__ xh,
    const _Float16* __restrict__ wqh, const _Float16* __restrict__ wkh,
    const _Float16* __restrict__ wvh,
    const float* __restrict__ v1, const float* __restrict__ plamb,
    float* __restrict__ qf, float* __restrict__ kf, _Float16* __restrict__ vT) {
  __shared__ __align__(16) _Float16 sA[SA_HALVES];
  __shared__ __align__(16) _Float16 sB[SB_HALVES];
  const int z = blockIdx.z;
  const _Float16* __restrict__ W = (z == 0) ? wqh : (z == 1) ? wkh : wvh;
  const int tid = threadIdx.x;
  const long row0 = (long)blockIdx.x * 64;
  const int  col0 = blockIdx.y * 128;

  v8f acc[4] = {};
  gemm_loop(sA, sB, xh + row0 * C_, W + (long)col0 * C_, tid, acc);

  const int lane = tid & 31, wid = tid >> 5;
  const int wm = wid & 3, wn = wid >> 2;
  const int n16 = lane & 15, grp = lane >> 4;
  const float lam = plamb[0];
#pragma unroll
  for (int j = 0; j < 4; ++j) {
#pragma unroll
    for (int r = 0; r < 8; ++r) {
      const long row = row0 + wm * 16 + r + grp * 8;
      const int  col = col0 + wn * 64 + j * 16 + n16;
      const float val = acc[j][r];
      if (z == 0) {
        qf[row * C_ + col] = val;
      } else if (z == 1) {
        kf[row * C_ + col] = val;
      } else {
        const float vv = (1.0f - lam) * val + lam * v1[row * C_ + col];
        const int b = (int)(row >> 11), t = (int)(row & (T_ - 1));
        const int h = col >> 6, d = col & 63;
        // V stored transposed: [B,H,D,T] so P@V B-fragments load contiguously
        vT[(((long)(b * H_ + h)) * D_ + d) * T_ + t] = (_Float16)vv;
      }
    }
  }
}

// ---------------- RMS-norm + rotary ----------------------
__global__ __launch_bounds__(256) void k_rmsrot(const float* __restrict__ in,
                                                _Float16* __restrict__ out) {
  const int lane = threadIdx.x & 31;
  const int w = threadIdx.x >> 5;
  const long row = (long)blockIdx.x * 8 + w;     // over B*T*H
  const int  h  = (int)(row % H_);
  const long bt = row / H_;
  const int  t  = (int)(bt % T_);
  const int  b  = (int)(bt / T_);

  const float* p = in + bt * C_ + h * D_;
  float x1 = p[lane], x2 = p[lane + 32];
  float ss = x1 * x1 + x2 * x2;
#pragma unroll
  for (int off = 16; off > 0; off >>= 1) ss += __shfl_xor(ss, off, 32);
  const float rms = rsqrtf(ss * (1.0f / 64.0f) + 1.1920928955078125e-07f);
  x1 *= rms; x2 *= rms;

  // inv_freq = 10000^(-lane/32) = exp(-lane * ln(1e4)/32)
  const float fr = (float)t * __expf(-(float)lane * 0.28782313662425574f);
  const float c = (float)(_Float16)__cosf(fr);   // fp16 rounding as in reference
  const float s = (float)(_Float16)__sinf(fr);

  _Float16* o = out + ((long)(b * H_ + h) * T_ + t) * D_;   // [B,H,T,D]
  o[lane]      = (_Float16)( x1 * c + x2 * s);
  o[lane + 32] = (_Float16)(-x1 * s + x2 * c);
}

// ---------------- flash attention ------------------------
// block 128 = 4 independent waves; wave = 16 query rows; key tiles of 32
__global__ __launch_bounds__(128) void k_attn(const _Float16* __restrict__ qh,
                                              const _Float16* __restrict__ kh,
                                              const _Float16* __restrict__ vT,
                                              _Float16* __restrict__ yh) {
  __shared__ __align__(16) _Float16 P[4][16][40];   // per-wave 16x32 P tile (+pad)
  const int lane = threadIdx.x & 31;
  const int w = threadIdx.x >> 5;
  const int n16 = lane & 15, grp = lane >> 4;
  const int bh    = blockIdx.x >> 5;     // b*H + h
  const int strip = blockIdx.x & 31;     // 32 strips of 64 rows
  const int qbase = strip * 64 + w * 16;

  const _Float16* q = qh + (long)bh * T_ * D_;
  const _Float16* k = kh + (long)bh * T_ * D_;
  const _Float16* v = vT + (long)bh * D_ * T_;

  const _Float16* qp = q + (long)(qbase + n16) * D_ + grp * 8;
  const v16h aq0 = comb(ldg8(qp),      ldg8(qp + 16));   // d 0..31
  const v16h aq1 = comb(ldg8(qp + 32), ldg8(qp + 48));   // d 32..63

  v8f ya[4] = {};
  float rmax[8], rsum[8];
#pragma unroll
  for (int r = 0; r < 8; ++r) { rmax[r] = -1e30f; rsum[r] = 0.0f; }
  const int qmax = qbase + 15;
  const v8f zz = {};

  for (int kc = 0; kc <= qmax; kc += 32) {
    const bool st1 = (kc + 16) <= qmax;
    // Batch all K fragment loads, pin, then the score WMMAs.
    const _Float16* kp0 = k + (long)(kc + n16) * D_ + grp * 8;
    v16h bk00 = comb(ldg8(kp0),      ldg8(kp0 + 16));
    v16h bk01 = comb(ldg8(kp0 + 32), ldg8(kp0 + 48));
    v16h bk10 = bk00, bk11 = bk01;
    if (st1) {
      const _Float16* kp1 = k + (long)(kc + 16 + n16) * D_ + grp * 8;
      bk10 = comb(ldg8(kp1),      ldg8(kp1 + 16));
      bk11 = comb(ldg8(kp1 + 32), ldg8(kp1 + 48));
    }
    asm volatile("" : "+v"(bk00), "+v"(bk01), "+v"(bk10), "+v"(bk11));
    v8f S0 = wmma_f16(aq0, bk00, zz);
    S0     = wmma_f16(aq1, bk01, S0);
    v8f S1 = zz;
    if (st1) {
      S1 = wmma_f16(aq0, bk10, zz);
      S1 = wmma_f16(aq1, bk11, S1);
    }

    // online softmax over the 32-key tile; rows live in lane-groups of 16
#pragma unroll
    for (int r = 0; r < 8; ++r) {
      const int m = r + grp * 8;
      const int qrow = qbase + m;
      float s0 = S0[r] * 0.125f;
      if (kc + n16 > qrow) s0 = -1e30f;
      float s1 = st1 ? S1[r] * 0.125f : -1e30f;
      if (kc + 16 + n16 > qrow) s1 = -1e30f;
      float mx = fmaxf(s0, s1);
#pragma unroll
      for (int off = 8; off > 0; off >>= 1) mx = fmaxf(mx, __shfl_xor(mx, off, 32));
      const float mnew  = fmaxf(rmax[r], mx);
      const float alpha = __expf(rmax[r] - mnew);
      rmax[r] = mnew;
      const float p0 = __expf(s0 - mnew);
      const float p1 = __expf(s1 - mnew);
      float ps = p0 + p1;
#pragma unroll
      for (int off = 8; off > 0; off >>= 1) ps += __shfl_xor(ps, off, 32);
      rsum[r] = rsum[r] * alpha + ps;
      ya[0][r] *= alpha; ya[1][r] *= alpha; ya[2][r] *= alpha; ya[3][r] *= alpha;
      P[w][m][n16]      = (_Float16)p0;   // accumulator-layout -> LDS
      P[w][m][16 + n16] = (_Float16)p1;
    }

    // reload P in A-fragment layout (same wave; DS in-order per wave)
    const h8 plo = *(const h8*)&P[w][n16][grp * 8];
    const h8 phi = *(const h8*)&P[w][n16][16 + grp * 8];
    v16h ap = comb(plo, phi);

    // y += P @ V: batch the 4 V fragments, pin, then 4 WMMAs
    const _Float16* vp = v + (long)n16 * T_ + kc + grp * 8;
    v16h bv0 = comb(ldg8(vp),                ldg8(vp + 16));
    v16h bv1 = comb(ldg8(vp + 16 * T_),      ldg8(vp + 16 * T_ + 16));
    v16h bv2 = comb(ldg8(vp + 32 * T_),      ldg8(vp + 32 * T_ + 16));
    v16h bv3 = comb(ldg8(vp + 48 * T_),      ldg8(vp + 48 * T_ + 16));
    asm volatile("" : "+v"(ap), "+v"(bv0), "+v"(bv1), "+v"(bv2), "+v"(bv3));
    ya[0] = wmma_f16(ap, bv0, ya[0]);
    ya[1] = wmma_f16(ap, bv1, ya[1]);
    ya[2] = wmma_f16(ap, bv2, ya[2]);
    ya[3] = wmma_f16(ap, bv3, ya[3]);
  }

  const int b = bh >> 4, h = bh & 15;
#pragma unroll
  for (int j = 0; j < 4; ++j) {
#pragma unroll
    for (int r = 0; r < 8; ++r) {
      const int m = r + grp * 8;
      yh[((long)(b * T_ + qbase + m)) * C_ + h * D_ + j * 16 + n16] =
          (_Float16)(ya[j][r] / rsum[r]);
    }
  }
}

// ---------------- output projection ----------------------
__global__ __launch_bounds__(256) void k_gemm_proj(const _Float16* __restrict__ yhm,
                                                   const _Float16* __restrict__ wph,
                                                   float* __restrict__ out) {
  __shared__ __align__(16) _Float16 sA[SA_HALVES];
  __shared__ __align__(16) _Float16 sB[SB_HALVES];
  const int tid = threadIdx.x;
  const long row0 = (long)blockIdx.x * 64;
  const int  col0 = blockIdx.y * 128;

  v8f acc[4] = {};
  gemm_loop(sA, sB, yhm + row0 * C_, wph + (long)col0 * C_, tid, acc);

  const int lane = tid & 31, wid = tid >> 5;
  const int wm = wid & 3, wn = wid >> 2;
  const int n16 = lane & 15, grp = lane >> 4;
#pragma unroll
  for (int j = 0; j < 4; ++j)
#pragma unroll
    for (int r = 0; r < 8; ++r)
      out[(row0 + wm * 16 + r + grp * 8) * C_ + col0 + wn * 64 + j * 16 + n16] =
          acc[j][r];
}

// ---------------- host launch ----------------------------
extern "C" void kernel_launch(void* const* d_in, const int* in_sizes, int n_in,
                              void* d_out, int out_size, void* d_ws, size_t ws_size,
                              hipStream_t stream) {
  (void)in_sizes; (void)n_in; (void)out_size; (void)ws_size;
  const float* x    = (const float*)d_in[0];
  const float* v1   = (const float*)d_in[1];
  const float* wq   = (const float*)d_in[2];
  const float* wk   = (const float*)d_in[3];
  const float* wv   = (const float*)d_in[4];
  const float* wp   = (const float*)d_in[5];
  const float* lamb = (const float*)d_in[6];
  float* out = (float*)d_out;

  char* ws = (char*)d_ws;
  size_t off = 0;
  _Float16* xh  = (_Float16*)(ws + off); off += (size_t)8192 * 1024 * 2;
  _Float16* wqh = (_Float16*)(ws + off); off += (size_t)1024 * 1024 * 2;
  _Float16* wkh = (_Float16*)(ws + off); off += (size_t)1024 * 1024 * 2;
  _Float16* wvh = (_Float16*)(ws + off); off += (size_t)1024 * 1024 * 2;
  _Float16* wph = (_Float16*)(ws + off); off += (size_t)1024 * 1024 * 2;
  const size_t qf_off = off;
  float* qf = (float*)(ws + off); off += (size_t)8192 * 1024 * 4;
  float* kf = (float*)(ws + off); off += (size_t)8192 * 1024 * 4;
  _Float16* vT  = (_Float16*)(ws + off); off += (size_t)8192 * 1024 * 2;
  _Float16* qhh = (_Float16*)(ws + off); off += (size_t)8192 * 1024 * 2;
  _Float16* khh = (_Float16*)(ws + off); off += (size_t)8192 * 1024 * 2;
  _Float16* yh  = (_Float16*)(ws + qf_off);   // reuse qf region after it is consumed

  k_cvt<<<32768, 256, 0, stream>>>(x,  xh,  8388608);
  k_cvt<<<4096,  256, 0, stream>>>(wq, wqh, 1048576);
  k_cvt<<<4096,  256, 0, stream>>>(wk, wkh, 1048576);
  k_cvt<<<4096,  256, 0, stream>>>(wv, wvh, 1048576);
  k_cvt<<<4096,  256, 0, stream>>>(wp, wph, 1048576);

  // second output of the tuple: v1 passthrough
  (void)hipMemcpyAsync(out + 8388608, v1, (size_t)8388608 * 4,
                       hipMemcpyDeviceToDevice, stream);

  k_gemm_qkv<<<dim3(128, 8, 3), 256, 0, stream>>>(xh, wqh, wkh, wvh, v1, lamb,
                                                  qf, kf, vT);
  k_rmsrot<<<16384, 256, 0, stream>>>(qf, qhh);
  k_rmsrot<<<16384, 256, 0, stream>>>(kf, khh);
  k_attn<<<2048, 128, 0, stream>>>(qhh, khh, vT, yh);
  k_gemm_proj<<<dim3(128, 8), 256, 0, stream>>>(yh, wph, out);
}
